// GeometricStructureEmbedding_77292231459089
// MI455X (gfx1250) — matrix operand order repaired and checked by reading, not compiled, not run on hardware
//
#include <hip/hip_runtime.h>
#include <hip/hip_bf16.h>

// GeometricStructureEmbedding for MI455X (gfx1250, wave32, WMMA).
//
// out[n,m,g] = (sinemb(dist(n,m)/0.2) @ Wd^T + bd)[g]
//            + max_k (sinemb(angle(n,m,k)*FACTOR_A) @ Wa^T + ba)[g]
//
// v_wmma_f32_16x16x32_f16 GEMM core. A fragments (sin/cos embedding rows) are
// synthesized directly in registers in the ISA 16-bit A-matrix lane layout.
// B fragments (Wd^T/Wa^T, f16) are staged in bank-padded LDS once per block
// and re-read per use via ds_load; an asm-laundered zero offset makes the
// addresses opaquely loop-variant so LICM can neither hoist them into 256
// VGPRs (round-2 failure) nor spill them to scratch (round-3 failure).

typedef __attribute__((ext_vector_type(16))) _Float16 v16h;
typedef __attribute__((ext_vector_type(8)))  _Float16 v8h;
typedef __attribute__((ext_vector_type(8)))  float    v8f;

#define HIDDEN 128
#define NPTS   512
#define KANG   8
#define ROWH   136   // padded LDS row stride in f16 (272B = 68 dwords)

// ------------------------------------------------- weights->f16 + freq table
__global__ void gse_setup_kernel(const float* __restrict__ Wd,
                                 const float* __restrict__ Wa,
                                 _Float16* __restrict__ Wd16,
                                 _Float16* __restrict__ Wa16,
                                 float* __restrict__ divt) {
  int i = blockIdx.x * blockDim.x + threadIdx.x;
  if (i < HIDDEN * HIDDEN) {
    Wd16[i] = (_Float16)Wd[i];
    Wa16[i] = (_Float16)Wa[i];
  }
  if (i < 64) {
    // div_term[j] = 10000^(-j/64)
    divt[i] = exp2f(-0.20762050593046015f * (float)i);
  }
}

// ---------------------------------------------------------------------- kNN(8)
__global__ void gse_knn_kernel(const float* __restrict__ pts,
                               int* __restrict__ knn) {
  int n = blockIdx.x * blockDim.x + threadIdx.x;
  if (n >= NPTS) return;
  float px = pts[n * 3 + 0], py = pts[n * 3 + 1], pz = pts[n * 3 + 2];
  float bestd[KANG + 1];
  int   besti[KANG + 1];
  for (int k = 0; k <= KANG; ++k) { bestd[k] = 3.4e38f; besti[k] = 0; }
  for (int m = 0; m < NPTS; ++m) {
    float dx = pts[m * 3 + 0] - px;
    float dy = pts[m * 3 + 1] - py;
    float dz = pts[m * 3 + 2] - pz;
    float d2 = dx * dx + dy * dy + dz * dz;
    if (d2 < bestd[KANG]) {
      int pos = KANG;
      while (pos > 0 && d2 < bestd[pos - 1]) {
        bestd[pos] = bestd[pos - 1];
        besti[pos] = besti[pos - 1];
        --pos;
      }
      bestd[pos] = d2;
      besti[pos] = m;
    }
  }
  // besti[0] is the point itself (d=0); keep the 8 true neighbors.
  for (int k = 0; k < KANG; ++k) knn[n * KANG + k] = besti[k + 1];
}

// -------------------------------------------------------- fused embed+GEMM+max
// One wave32 per 16-row tile of the (512*512)x128 output. Tile t covers
// n = t/32, m in [ (t%32)*16, +16 ). 288 dynamic WMMAs per wave.
__global__ __launch_bounds__(256) void gse_main_kernel(
    const float* __restrict__ pts,
    const int* __restrict__ knn,
    const _Float16* __restrict__ Wd16,
    const _Float16* __restrict__ Wa16,
    const float* __restrict__ bd,
    const float* __restrict__ ba,
    const float* __restrict__ divt,
    float* __restrict__ out) {
  extern __shared__ _Float16 ldsW[];   // rows 0-127: Wd, rows 128-255: Wa

  // ---- cooperative fill: 256 rows x 16 chunks of 16B, bank-padded rows.
  for (int idx = threadIdx.x; idx < 256 * 16; idx += 256) {
    const int r = idx >> 4, c = idx & 15;
    const _Float16* src = (r < HIDDEN)
                              ? (Wd16 + (size_t)r * HIDDEN + c * 8)
                              : (Wa16 + (size_t)(r - HIDDEN) * HIDDEN + c * 8);
    *(v8h*)(ldsW + r * ROWH + c * 8) = *(const v8h*)src;
  }
  __syncthreads();

  const int lane  = threadIdx.x & 31;
  const int wave  = threadIdx.x >> 5;
  const int tile  = blockIdx.x * 8 + wave;        // 0 .. 16383
  const int n     = tile >> 5;                    // shared by all 16 rows
  const int mbase = (tile & 31) << 4;
  const int lm    = lane & 15;
  const int hi    = (lane >> 4) & 1;              // upper half-wave
  const int m     = mbase + lm;                   // this lane's output row

  // Geometry (pn is wave-uniform; pm per lane, duplicated across half-waves).
  const float pnx = pts[n * 3 + 0], pny = pts[n * 3 + 1], pnz = pts[n * 3 + 2];
  const float pmx = pts[m * 3 + 0], pmy = pts[m * 3 + 1], pmz = pts[m * 3 + 2];
  const float ax = pmx - pnx, ay = pmy - pny, az = pmz - pnz;
  const float dist = sqrtf(ax * ax + ay * ay + az * az);
  const float td = dist * 5.0f;                   // dist / SIGMA_D

  // Frequency table for exactly the h-indices this lane owns in the A layout.
  // A 16x32 f16 fragment, K window kt*32: lanes 0-15 own h = {0..7,16..23},
  // lanes 16-31 own h = {8..15,24..31}. h=2j -> sin(t*div[j]), h=2j+1 -> cos.
  float divv[32];
#pragma unroll
  for (int kt = 0; kt < 4; ++kt) {
#pragma unroll
    for (int c = 0; c < 8; ++c) {
      const int j = kt * 16 + ((c < 4) ? c : (c + 4)) + hi * 4;
      divv[kt * 8 + c] = divt[j];
    }
  }

  v16h frag[4];
  auto build_frags = [&](float t) {
#pragma unroll
    for (int kt = 0; kt < 4; ++kt) {
      v16h f;
#pragma unroll
      for (int c = 0; c < 8; ++c) {
        const float w = t * divv[kt * 8 + c];
        f[2 * c + 0] = (_Float16)__sinf(w);
        f[2 * c + 1] = (_Float16)__cosf(w);
      }
      frag[kt] = f;
    }
  };

  // B fragment (32x16 KxN = W^T tile) from padded LDS: lane holds column
  // g = nt*16+lm, h at kt*32+hi*8+{0..7} and kt*32+16+hi*8+{0..7}.
  // Two 16B ds loads (272B row stride is 16B-aligned only).
  const int bcol = hi * 8;
  auto loadB = [&](int rowbase, unsigned lnd, int nt, int kt) -> v16h {
    const _Float16* p =
        ldsW + (unsigned)(rowbase + nt * 16 + lm) * ROWH + kt * 32 + bcol + lnd;
    v8h lo = *(const v8h*)p;
    v8h hh = *(const v8h*)(p + 16);
    v16h b;
#pragma unroll
    for (int e = 0; e < 8; ++e) { b[e] = lo[e]; b[e + 8] = hh[e]; }
    return b;
  };

  const v8f vzero = {0.f, 0.f, 0.f, 0.f, 0.f, 0.f, 0.f, 0.f};

  // ---- angle branch FIRST: max over 8 neighbors of emb_a @ Wa^T.
  // Only accm[8] (64 VGPRs) persists through the hot loop. kk rolled; the
  // asm-laundered zero offset keeps all B ds_loads inside each iteration.
  v8f accm[8];
  const v8f vninf = {-3.0e38f, -3.0e38f, -3.0e38f, -3.0e38f,
                     -3.0e38f, -3.0e38f, -3.0e38f, -3.0e38f};
#pragma unroll
  for (int nt = 0; nt < 8; ++nt) accm[nt] = vninf;

  unsigned zoff = 0;   // always 0, but opaque to the optimizer
#pragma unroll 1
  for (int kk = 0; kk < KANG; ++kk) {
    asm volatile("" : "+v"(zoff));           // block LICM/CSE across iterations
    const int j = knn[n * KANG + kk];        // uniform scalar load
    const float rx = pts[j * 3 + 0] - pnx;
    const float ry = pts[j * 3 + 1] - pny;
    const float rz = pts[j * 3 + 2] - pnz;
    const float cx = ry * az - rz * ay;
    const float cy = rz * ax - rx * az;
    const float cz = rx * ay - ry * ax;
    const float sv = sqrtf(cx * cx + cy * cy + cz * cz);
    const float cv = rx * ax + ry * ay + rz * az;
    const float ta = atan2f(sv, cv) * 3.8197186342054885f;  // 180/(15*pi)
    build_frags(ta);
#pragma unroll
    for (int nt = 0; nt < 8; ++nt) {
      v8f acc = vzero;
#pragma unroll
      for (int kt = 0; kt < 4; ++kt) {
        v16h b = loadB(HIDDEN, zoff, nt, kt);   // Wa rows live at 128..255
        acc = __builtin_amdgcn_wmma_f32_16x16x32_f16(false, frag[kt], false, b,
                                                     (short)0, acc, false, false);
      }
#pragma unroll
      for (int e = 0; e < 8; ++e) accm[nt][e] = fmaxf(accm[nt][e], acc[e]);
    }
  }

  // ---- distance branch, fused with bias-add + store (transient accumulator).
  // C/D layout: VGPR e, lane -> (M = e + hi*8, N = lm).
  build_frags(td);
  float* orow = out + (size_t)(n * NPTS + mbase) * HIDDEN;
  const int mofs = hi * 8;
#pragma unroll
  for (int nt = 0; nt < 8; ++nt) {
    v8f acc = vzero;
#pragma unroll
    for (int kt = 0; kt < 4; ++kt) {
      v16h b = loadB(0, 0u, nt, kt);            // Wd rows live at 0..127
      acc = __builtin_amdgcn_wmma_f32_16x16x32_f16(false, frag[kt], false, b,
                                                   (short)0, acc, false, false);
    }
    const int g = nt * 16 + lm;
    const float bsum = bd[g] + ba[g];
#pragma unroll
    for (int e = 0; e < 8; ++e) {
      orow[(size_t)(e + mofs) * HIDDEN + g] = acc[e] + accm[nt][e] + bsum;
    }
  }
}

// ------------------------------------------------------------------- launcher
extern "C" void kernel_launch(void* const* d_in, const int* in_sizes, int n_in,
                              void* d_out, int out_size, void* d_ws, size_t ws_size,
                              hipStream_t stream) {
  (void)in_sizes; (void)n_in; (void)out_size; (void)ws_size;
  const float* pts = (const float*)d_in[0];
  const float* Wd  = (const float*)d_in[1];
  const float* bd  = (const float*)d_in[2];
  const float* Wa  = (const float*)d_in[3];
  const float* ba  = (const float*)d_in[4];
  float* out = (float*)d_out;

  // workspace: Wd16 (32KB) | Wa16 (32KB) | knn idx (16KB) | div table (256B)
  _Float16* Wd16 = (_Float16*)d_ws;
  _Float16* Wa16 = Wd16 + HIDDEN * HIDDEN;
  int*      knn  = (int*)(Wa16 + HIDDEN * HIDDEN);
  float*    divt = (float*)(knn + NPTS * KANG);

  gse_setup_kernel<<<(HIDDEN * HIDDEN + 255) / 256, 256, 0, stream>>>(
      Wd, Wa, Wd16, Wa16, divt);
  gse_knn_kernel<<<(NPTS + 255) / 256, 256, 0, stream>>>(pts, knn);

  // 512*512/16 = 16384 tiles, 8 waves (256 threads) per block.
  // Dynamic LDS: 256 padded rows of 272B = 69632B.
  const size_t lds_bytes = 256 * ROWH * sizeof(_Float16);
  gse_main_kernel<<<16384 / 8, 256, lds_bytes, stream>>>(
      pts, knn, Wd16, Wa16, bd, ba, divt, out);
}
